// FFSpikingNet_62723702391150
// MI455X (gfx1250) — matrix operand order repaired and checked by reading, not compiled, not run on hardware
//
#include <hip/hip_runtime.h>
#include <hip/hip_bf16.h>

typedef _Float16 v8h  __attribute__((ext_vector_type(8)));
typedef _Float16 v16h __attribute__((ext_vector_type(16)));
typedef float    v8f  __attribute__((ext_vector_type(8)));

#define T_STEPS 100
#define BATCH   256
#define F_IN    720
#define F_PAD   736   // 720 padded up to multiple of 32
#define HID     2048
#define MROWS   (T_STEPS * BATCH)   // 25600

// ---------------------------------------------------------------------------
// f32 -> f16 convert with optional K padding (pad region zero-filled)
// ---------------------------------------------------------------------------
__global__ void cvt_f32_to_f16_pad(const float* __restrict__ src,
                                   _Float16* __restrict__ dst,
                                   int rows, int kSrc, int kDst) {
    size_t idx = (size_t)blockIdx.x * blockDim.x + threadIdx.x;
    size_t total = (size_t)rows * kDst;
    if (idx >= total) return;
    int r = (int)(idx / kDst);
    int k = (int)(idx % kDst);
    dst[idx] = (k < kSrc) ? (_Float16)src[(size_t)r * kSrc + k] : (_Float16)0.0f;
}

// ---------------------------------------------------------------------------
// WMMA GEMM:  C[m,n] = sum_k A[m,k] * W[n,k] + bias[n]
//   A: [M x K] f16 row-major (lda), W: [N x K] f16 row-major (ldb), C: f32 [M x N]
//   Block tile 128x256, 8 waves (wave32), wave tile 64x64 (4x4 WMMA 16x16 tiles)
//   -> 16 b128 loads feed 16 WMMAs per K-step (2x register reuse vs 64x32 tile)
//   Grid: (N/256, M/128)
// ---------------------------------------------------------------------------
__global__ __launch_bounds__(256)
void gemm_f16_wmma(const _Float16* __restrict__ A, int lda,
                   const _Float16* __restrict__ W, int ldb,
                   const float* __restrict__ bias,
                   float* __restrict__ C, int N, int K) {
    const int lane = threadIdx.x & 31;
    const int wid  = threadIdx.x >> 5;
    const int l16  = lane & 15;       // 0..15
    const int hi   = lane >> 4;       // 0 or 1

    const int blockM = blockIdx.y * 128;
    const int blockN = blockIdx.x * 256;
    const int waveM  = blockM + (wid & 1) * 64;   // 2 waves along M
    const int waveN  = blockN + (wid >> 1) * 64;  // 4 waves along N

    v8f acc[4][4];
#pragma unroll
    for (int i = 0; i < 4; ++i)
#pragma unroll
        for (int j = 0; j < 4; ++j)
#pragma unroll
            for (int r = 0; r < 8; ++r) acc[i][j][r] = 0.0f;

    // Per-lane base pointers.
    // A fragment (contiguous k order, same as B): lane row = tileM + l16,
    //   lanes 0-15: K=k0..k0+15; lanes 16-31: +16.
    // B fragment: lane col = tileN + l16, lanes 0-15: K=k0..k0+15, lanes 16-31: +16.
    const _Float16* Arow[4];
#pragma unroll
    for (int i = 0; i < 4; ++i)
        Arow[i] = A + (size_t)(waveM + i * 16 + l16) * lda + hi * 16;
    const _Float16* Wrow[4];
#pragma unroll
    for (int j = 0; j < 4; ++j)
        Wrow[j] = W + (size_t)(waveN + j * 16 + l16) * ldb + hi * 16;

    for (int k0 = 0; k0 < K; k0 += 32) {
        // prefetch next K-step (emits global_prefetch_b8)
        if (k0 + 32 < K) {
            __builtin_prefetch(Arow[0] + k0 + 32, 0, 1);
            __builtin_prefetch(Wrow[0] + k0 + 32, 0, 1);
        }

        v16h afrag[4], bfrag[4];
#pragma unroll
        for (int i = 0; i < 4; ++i) {
            v8h a0 = *reinterpret_cast<const v8h*>(Arow[i] + k0);
            v8h a1 = *reinterpret_cast<const v8h*>(Arow[i] + k0 + 8);
            afrag[i] = __builtin_shufflevector(a0, a1,
                0, 1, 2, 3, 4, 5, 6, 7, 8, 9, 10, 11, 12, 13, 14, 15);
        }
#pragma unroll
        for (int j = 0; j < 4; ++j) {
            v8h b0 = *reinterpret_cast<const v8h*>(Wrow[j] + k0);
            v8h b1 = *reinterpret_cast<const v8h*>(Wrow[j] + k0 + 8);
            bfrag[j] = __builtin_shufflevector(b0, b1,
                0, 1, 2, 3, 4, 5, 6, 7, 8, 9, 10, 11, 12, 13, 14, 15);
        }
#pragma unroll
        for (int i = 0; i < 4; ++i)
#pragma unroll
            for (int j = 0; j < 4; ++j)
                acc[i][j] = __builtin_amdgcn_wmma_f32_16x16x32_f16(
                    false, afrag[i], false, bfrag[j],
                    (short)0, acc[i][j], false, false);
    }

    // Epilogue: C/D layout — VGPR r holds row r (lanes 0-15) / row r+8 (lanes 16-31),
    // column = tile col + lane%16.
#pragma unroll
    for (int j = 0; j < 4; ++j) {
        const int col = waveN + j * 16 + l16;
        const float bval = bias[col];
#pragma unroll
        for (int i = 0; i < 4; ++i) {
            const int row0 = waveM + i * 16 + hi * 8;
#pragma unroll
            for (int r = 0; r < 8; ++r)
                C[(size_t)(row0 + r) * N + col] = acc[i][j][r] + bval;
        }
    }
}

// ---------------------------------------------------------------------------
// Row-wise L2 normalize: cur[row, :] /= max(||cur[row,:]||_2, 1e-12)
// One block (256 threads) per row of length H.
// ---------------------------------------------------------------------------
__global__ __launch_bounds__(256)
void row_l2_normalize(float* __restrict__ cur, int H) {
    __shared__ float red[256];
    float* row = cur + (size_t)blockIdx.x * H;
    float ss = 0.0f;
    for (int h = threadIdx.x; h < H; h += 256) {
        float v = row[h];
        ss += v * v;
    }
    red[threadIdx.x] = ss;
    __syncthreads();
    for (int s = 128; s > 0; s >>= 1) {
        if ((int)threadIdx.x < s) red[threadIdx.x] += red[threadIdx.x + s];
        __syncthreads();
    }
    float scale = 1.0f / fmaxf(sqrtf(red[0]), 1e-12f);
    for (int h = threadIdx.x; h < H; h += 256)
        row[h] *= scale;
}

// ---------------------------------------------------------------------------
// LIF scan: v += (c - v)/tau (tau=2 -> v = 0.5*(v+c)), spike at v>=1, hard
// reset to 0. One thread per (b,h); sequential over t. Emits f16 spikes for
// the next layer's GEMM A matrix and float spike counts.
// ---------------------------------------------------------------------------
__global__ __launch_bounds__(256)
void lif_scan(const float* __restrict__ cur, _Float16* __restrict__ spk,
              float* __restrict__ counts, size_t BH, int store_spikes) {
    size_t idx = (size_t)blockIdx.x * blockDim.x + threadIdx.x;
    if (idx >= BH) return;
    float v = 0.0f, cnt = 0.0f;
    for (int t = 0; t < T_STEPS; ++t) {
        float c = cur[(size_t)t * BH + idx];
        v = 0.5f * (v + c);
        float s = (v >= 1.0f) ? 1.0f : 0.0f;
        cnt += s;
        v = v * (1.0f - s);
        if (store_spikes) spk[(size_t)t * BH + idx] = (_Float16)s;
    }
    counts[idx] = cnt;
}

// ---------------------------------------------------------------------------
// Goodness: out[layer, b] = mean_h (counts[b,h]/T)^2. One block per b.
// ---------------------------------------------------------------------------
__global__ __launch_bounds__(256)
void goodness(const float* __restrict__ counts, float* __restrict__ out,
              int layer) {
    __shared__ float red[256];
    const float* c = counts + (size_t)blockIdx.x * HID;
    float ss = 0.0f;
    const float invT = 1.0f / (float)T_STEPS;
    for (int h = threadIdx.x; h < HID; h += 256) {
        float r = c[h] * invT;
        ss += r * r;
    }
    red[threadIdx.x] = ss;
    __syncthreads();
    for (int s = 128; s > 0; s >>= 1) {
        if ((int)threadIdx.x < s) red[threadIdx.x] += red[threadIdx.x + s];
        __syncthreads();
    }
    if (threadIdx.x == 0)
        out[(size_t)layer * BATCH + blockIdx.x] = red[0] / (float)HID;
}

// ---------------------------------------------------------------------------
extern "C" void kernel_launch(void* const* d_in, const int* in_sizes, int n_in,
                              void* d_out, int out_size, void* d_ws, size_t ws_size,
                              hipStream_t stream) {
    (void)in_sizes; (void)n_in; (void)out_size; (void)ws_size;

    const float* x  = (const float*)d_in[0];  // [T,B,720]
    const float* W0 = (const float*)d_in[1];  // [2048,720]
    const float* b0 = (const float*)d_in[2];
    const float* W1 = (const float*)d_in[3];  // [2048,2048]
    const float* b1 = (const float*)d_in[4];
    const float* W2 = (const float*)d_in[5];
    const float* b2 = (const float*)d_in[6];
    float* out = (float*)d_out;               // [3,256]

    char* ws = (char*)d_ws;
    size_t off = 0;
    auto carve = [&](size_t bytes) -> void* {
        void* p = ws + off;
        off = (off + bytes + 255) & ~(size_t)255;
        return p;
    };

    _Float16* xh     = (_Float16*)carve((size_t)MROWS * F_PAD * 2);
    _Float16* W0h    = (_Float16*)carve((size_t)HID * F_PAD * 2);
    _Float16* W1h    = (_Float16*)carve((size_t)HID * HID * 2);
    _Float16* W2h    = (_Float16*)carve((size_t)HID * HID * 2);
    float*    cur    = (float*)   carve((size_t)MROWS * HID * 4);
    _Float16* spkA   = (_Float16*)carve((size_t)MROWS * HID * 2);
    _Float16* spkB   = (_Float16*)carve((size_t)MROWS * HID * 2);
    float*    counts = (float*)   carve((size_t)BATCH * HID * 4);

    const size_t BH = (size_t)BATCH * HID;  // 524288
    dim3 blk(256);

    // --- convert inputs / weights to f16 (x, W0 padded K 720->736) ---
    {
        size_t n;
        n = (size_t)MROWS * F_PAD;
        cvt_f32_to_f16_pad<<<dim3((n + 255) / 256), blk, 0, stream>>>(x, xh, MROWS, F_IN, F_PAD);
        n = (size_t)HID * F_PAD;
        cvt_f32_to_f16_pad<<<dim3((n + 255) / 256), blk, 0, stream>>>(W0, W0h, HID, F_IN, F_PAD);
        n = (size_t)HID * HID;
        cvt_f32_to_f16_pad<<<dim3((n + 255) / 256), blk, 0, stream>>>(W1, W1h, HID, HID, HID);
        cvt_f32_to_f16_pad<<<dim3((n + 255) / 256), blk, 0, stream>>>(W2, W2h, HID, HID, HID);
    }

    dim3 gemm_grid(HID / 256, MROWS / 128);   // (8, 200)
    dim3 lif_grid((unsigned)(BH / 256));      // 2048
    dim3 norm_grid(MROWS);                    // 25600 rows
    dim3 good_grid(BATCH);                    // 256

    // ---------------- Layer 0 ----------------
    gemm_f16_wmma<<<gemm_grid, blk, 0, stream>>>(xh, F_PAD, W0h, F_PAD, b0, cur, HID, F_PAD);
    lif_scan<<<lif_grid, blk, 0, stream>>>(cur, spkA, counts, BH, 1);
    goodness<<<good_grid, blk, 0, stream>>>(counts, out, 0);

    // ---------------- Layer 1 ----------------
    gemm_f16_wmma<<<gemm_grid, blk, 0, stream>>>(spkA, HID, W1h, HID, b1, cur, HID, HID);
    row_l2_normalize<<<norm_grid, blk, 0, stream>>>(cur, HID);
    lif_scan<<<lif_grid, blk, 0, stream>>>(cur, spkB, counts, BH, 1);
    goodness<<<good_grid, blk, 0, stream>>>(counts, out, 1);

    // ---------------- Layer 2 ----------------
    gemm_f16_wmma<<<gemm_grid, blk, 0, stream>>>(spkB, HID, W2h, HID, b2, cur, HID, HID);
    row_l2_normalize<<<norm_grid, blk, 0, stream>>>(cur, HID);
    lif_scan<<<lif_grid, blk, 0, stream>>>(cur, nullptr, counts, BH, 0);
    goodness<<<good_grid, blk, 0, stream>>>(counts, out, 2);
}
